// ESIM_3461743640651
// MI455X (gfx1250) — matrix operand order repaired and checked by reading, not compiled, not run on hardware
//
#include <hip/hip_runtime.h>
#include <hip/hip_bf16.h>

// ---------------------------------------------------------------------------
// ESIM forward on gfx1250 (CDNA5). All large GEMMs run on
// v_wmma_f32_16x16x32_f16 (f32 accumulate, f16 operands staged through LDS in
// fragment order so fragment loads are 2x ds_load_b128 per lane, and staging
// is double-buffered so global loads overlap the WMMAs).
// Shapes: B=64, L=64, T=63, S=1+L+T=128, D=300, CF=100, DE=400, U=300,
// H4=1200, BB=2*B=128 (both sentences batched through shared encoder).
// ---------------------------------------------------------------------------

#define BQ 64
#define LQ 64
#define TQ 63
#define SQ 128          // 1 + L + T
#define DQ 300
#define CFQ 100
#define DEQ 400         // D + CF
#define UQ 300
#define H4Q 1200
#define BBQ 128         // 2*B
#define NCQ 3

typedef __attribute__((ext_vector_type(16))) _Float16 v16h;
typedef __attribute__((ext_vector_type(2)))  _Float16 v2h;
typedef __attribute__((ext_vector_type(8)))  float    v8f;

// ------------------------------ GEMM modes ---------------------------------
// AM: 0 = plain A[b*sAb + m*ldA + k]
//     1 = K-concat: k<K1 -> A[m*ldA+k] else A2[m*ldA2 + (k-K1)]
//     2 = mfuse: seg=k/K1, j=k%K1; [h, att, h*att, h-att] from A(=h), A2(=att)
// BM: 0 = plain B[b*sBb + k*ldB + n]
//     1 = K-concat: k<KB1 -> B[k*ldB+n] else B2[(k-KB1)*ldB2+n]
//     2 = transposed B[b*sBb + n*ldB + k]
// ACT: 0 none, 1 relu, 2 tanh
struct GemmP {
  const float* A;  long ldA;  long sAb;
  const float* A2; long ldA2;
  int K1;
  const float* Bm; long ldB;  long sBb;
  const float* B2; long ldB2;
  int KB1;
  const float* bias;
  float* C; long ldC; long sCb;
  int M, N, K;
};

template<int AM>
__device__ __forceinline__ float fetchA(const GemmP& p, int b, int m, int k) {
  if (AM == 0) return p.A[(long)b * p.sAb + (long)m * p.ldA + k];
  if (AM == 1) {
    return (k < p.K1) ? p.A [(long)m * p.ldA  + k]
                      : p.A2[(long)m * p.ldA2 + (k - p.K1)];
  }
  // AM == 2 : mfuse
  int seg = k / p.K1;
  int j   = k - seg * p.K1;
  float hv = p.A [(long)m * p.ldA  + j];
  float av = p.A2[(long)m * p.ldA2 + j];
  return (seg == 0) ? hv : (seg == 1) ? av : (seg == 2) ? hv * av : hv - av;
}

template<int BM>
__device__ __forceinline__ float fetchB(const GemmP& p, int b, int k, int n) {
  if (BM == 0) return p.Bm[(long)b * p.sBb + (long)k * p.ldB + n];
  if (BM == 1) {
    return (k < p.KB1) ? p.Bm[(long)k * p.ldB + n]
                       : p.B2[(long)(k - p.KB1) * p.ldB2 + n];
  }
  return p.Bm[(long)b * p.sBb + (long)n * p.ldB + k];   // transposed
}

// Block = 128 threads = 4 waves (wave32). Block tile = 64x64 C, wave = 16x64.
// LDS tiles are stored in WMMA *fragment order*:
//   A buffer: [wave][lane][16 halfs]  (lane's v16h is 32 contiguous bytes)
//   B buffer: [nb]  [lane][16 halfs]
// Inverse maps used at staging time (contiguous k order, same for A and B):
//   A (m,k): lane = 16*(k>>4) + (m&15),     half = k&15
//   B (k,n): lane = 16*(k>>4) + (n&15),     half = k&15
template<int AM, int BM, int ACT>
__global__ __launch_bounds__(128)
void gemm_wmma_k(GemmP p) {
  __shared__ __align__(32) _Float16 sA[2][2048];
  __shared__ __align__(32) _Float16 sB[2][2048];
  const int tid  = threadIdx.x;
  const int lane = tid & 31;
  const int wave = tid >> 5;
  const int bz   = blockIdx.z;
  const int tM   = blockIdx.y * 64;
  const int tN   = blockIdx.x * 64;

  v8f acc[4] = {};

  const int lm = lane & 15;

  // Stage one 64x32 A chunk (fragment order) starting at k-offset kbase.
  auto stageA = [&](int kbase, _Float16* dA) {
#pragma unroll
    for (int i = 0; i < 8; ++i) {
      int dh     = (tid + i * 128) * 2;        // even dest half index
      int w      = dh >> 9;                    // target wave fragment
      int lane_d = (dh >> 4) & 31;
      int h0     = dh & 15;                    // even
      int m      = tM + w * 16 + (lane_d & 15);
      int lhd    = lane_d >> 4;
      int ka     = h0 + 16 * lhd;
      int kg0    = kbase + ka;                 // h0 even => h0+1 maps to kg0+1
      float a0 = 0.0f, a1 = 0.0f;
      if (m < p.M) {
        if (kg0     < p.K) a0 = fetchA<AM>(p, bz, m, kg0);
        if (kg0 + 1 < p.K) a1 = fetchA<AM>(p, bz, m, kg0 + 1);
      }
      *(v2h*)&dA[dh] = (v2h){(_Float16)a0, (_Float16)a1};
    }
  };

  // Stage one 32x64 B chunk (fragment order).
  auto stageB = [&](int kbase, _Float16* dB) {
#pragma unroll
    for (int i = 0; i < 8; ++i) {
      int dh     = (tid + i * 128) * 2;
      int nb     = dh >> 9;                    // target column block
      int lane_d = (dh >> 4) & 31;
      int h0     = dh & 15;
      int n      = tN + nb * 16 + (lane_d & 15);
      int kg0    = kbase + (lane_d >> 4) * 16 + h0;
      float b0 = 0.0f, b1 = 0.0f;
      if (n < p.N) {
        if (kg0     < p.K) b0 = fetchB<BM>(p, bz, kg0, n);
        if (kg0 + 1 < p.K) b1 = fetchB<BM>(p, bz, kg0 + 1, n);
      }
      *(v2h*)&dB[dh] = (v2h){(_Float16)b0, (_Float16)b1};
    }
  };

  const int nChunks = (p.K + 31) / 32;
  stageA(0, sA[0]);
  stageB(0, sB[0]);
  __syncthreads();

  int cur = 0;
  for (int c = 0; c < nChunks; ++c) {
    int nxt = cur ^ 1;
    if (c + 1 < nChunks) {       // prefetch next chunk while computing current
      stageA((c + 1) * 32, sA[nxt]);
      stageB((c + 1) * 32, sB[nxt]);
    }
    v16h af = *(const v16h*)&sA[cur][(wave * 32 + lane) * 16];
#pragma unroll
    for (int nb = 0; nb < 4; ++nb) {
      v16h bf = *(const v16h*)&sB[cur][(nb * 32 + lane) * 16];
      acc[nb] = __builtin_amdgcn_wmma_f32_16x16x32_f16(
          false, af, false, bf, (short)0, acc[nb], false, false);
    }
    __syncthreads();   // staging of nxt done; compute of cur done
    cur = nxt;
  }

  // Epilogue: C layout — lane n = lane&15 (+16*nb), VGPR r -> M = 8*(lane>=16)+r
#pragma unroll
  for (int nb = 0; nb < 4; ++nb) {
    int n = tN + nb * 16 + lm;
    if (n >= p.N) continue;
    float bv = p.bias ? p.bias[n] : 0.0f;
#pragma unroll
    for (int r = 0; r < 8; ++r) {
      int m = tM + wave * 16 + (lane >> 4) * 8 + r;
      if (m >= p.M) continue;
      float v = acc[nb][r] + bv;
      if (ACT == 1) v = fmaxf(v, 0.0f);
      if (ACT == 2) v = tanhf(v);
      p.C[(long)bz * p.sCb + (long)m * p.ldC + n] = v;
    }
  }
}

template<int AM, int BM, int ACT>
static inline void run_gemm(hipStream_t st, const GemmP& p, int batches) {
  dim3 grid((unsigned)((p.N + 63) / 64), (unsigned)((p.M + 63) / 64),
            (unsigned)batches);
  gemm_wmma_k<AM, BM, ACT><<<grid, dim3(128), 0, st>>>(p);
}

// ------------------------------ small kernels ------------------------------

__global__ void k_zero(float* p, long n) {
  long i = (long)blockIdx.x * blockDim.x + threadIdx.x;
  if (i < n) p[i] = 0.0f;
}

// tree[r, 1+l, 0:300] = word_embed[x[b,l]]
__global__ void k_embed(const int* x1, const int* x2, const float* wemb,
                        float* tree) {
  long idx = (long)blockIdx.x * blockDim.x + threadIdx.x;
  if (idx >= (long)BBQ * LQ * DQ) return;
  int d = (int)(idx % DQ);
  long w = idx / DQ;
  int l = (int)(w % LQ);
  int r = (int)(w / LQ);
  int b = r & 63, sent = r >> 6;
  int tok = (sent ? x2 : x1)[b * LQ + l];
  tree[((long)r * SQ + (1 + l)) * DEQ + d] = wemb[(long)tok * DQ + d];
}

// char CNN: embed 16 chars (CE=8), conv window 5 VALID (12 pos), CF=100, max.
__global__ void k_char(const int* c1, const int* c2, const float* cemb,
                       const float* cfilt, float* tree) {
  int w = blockIdx.x;                       // 0..8191
  int l = w & 63, r = w >> 6;
  int b = r & 63, sent = r >> 6;
  const int* cp = (sent ? c2 : c1) + ((long)b * LQ + l) * 16;
  __shared__ float e[16][8];
  __shared__ float filt[5 * 8 * CFQ];       // (w, c, f) row-major, 16 KB
  for (int i = threadIdx.x; i < 16 * 8; i += blockDim.x) {
    int pos = i >> 3, c = i & 7;
    e[pos][c] = cemb[cp[pos] * 8 + c];
  }
  for (int i = threadIdx.x; i < 5 * 8 * CFQ; i += blockDim.x) filt[i] = cfilt[i];
  __syncthreads();
  for (int f = threadIdx.x; f < CFQ; f += blockDim.x) {
    float mx = -1e30f;
    for (int pos = 0; pos < 12; ++pos) {
      float s = 0.0f;
#pragma unroll
      for (int wv = 0; wv < 5; ++wv)
#pragma unroll
        for (int c = 0; c < 8; ++c)
          s += e[pos + wv][c] * filt[(wv * 8 + c) * CFQ + f];
      mx = fmaxf(mx, s);
    }
    tree[((long)r * SQ + 1 + l) * DEQ + DQ + f] = mx;
  }
}

// tree unfold: per (sent,b) sequential over i=1..62; per-dim chains independent.
__global__ void k_unfold(const int* t1, const int* t2, const int* l1,
                         const int* l2, float* tree) {
  int r = blockIdx.x;                       // 0..127
  int b = r & 63, sent = r >> 6;
  const int* tp = (sent ? t2 : t1) + (long)b * TQ * 2;
  int len = (sent ? l2 : l1)[b];
  float* tr = tree + (long)r * SQ * DEQ;
  for (int i = 1; i < TQ; ++i) {
    int c0 = tp[i * 2 + 0], c1 = tp[i * 2 + 1];
    int tgt = len + 1 + i;
    for (int d = threadIdx.x; d < DEQ; d += blockDim.x) {
      float pv = 0.5f * (tr[(long)c0 * DEQ + d] + tr[(long)c1 * DEQ + d]);
      tr[(long)tgt * DEQ + d] += pv;
    }
  }
}

__device__ __forceinline__ float sigm(float x) {
  return 1.0f / (1.0f + expf(-x));
}

// LSTM cell: gates from R(128,1200)+bias; update h,c; write seq[:,t,colOff:+300]
__global__ void k_cell(const float* R, const float* bias, float* hs, float* cs,
                       float* seq, int t, int colOff) {
  int idx = blockIdx.x * blockDim.x + threadIdx.x;
  if (idx >= BBQ * UQ) return;
  int r = idx / UQ, u = idx - r * UQ;
  const float* Rr = R + (long)r * H4Q;
  float gi = Rr[u]           + bias[u];
  float gg = Rr[UQ + u]      + bias[UQ + u];
  float gf = Rr[2 * UQ + u]  + bias[2 * UQ + u];
  float go = Rr[3 * UQ + u]  + bias[3 * UQ + u];
  float c = cs[idx];
  c = sigm(gf + 1.0f) * c + sigm(gi) * tanhf(gg);
  float h = sigm(go) * tanhf(c);
  cs[idx] = c;
  hs[idx] = h;
  seq[((long)r * SQ + t) * 600 + colOff + u] = h;
}

// softmax over last axis of sim[b,s,:] -> P[b,s,:]
__global__ void k_softmax_rows(const float* sim, float* P) {
  int b = blockIdx.y, s = blockIdx.x, tid = threadIdx.x;   // 128 threads
  float v = sim[((long)b * SQ + s) * SQ + tid];
  __shared__ float red[SQ];
  red[tid] = v; __syncthreads();
  for (int o = 64; o; o >>= 1) { if (tid < o) red[tid] = fmaxf(red[tid], red[tid + o]); __syncthreads(); }
  float mx = red[0]; __syncthreads();
  float e = expf(v - mx);
  red[tid] = e; __syncthreads();
  for (int o = 64; o; o >>= 1) { if (tid < o) red[tid] += red[tid + o]; __syncthreads(); }
  P[((long)b * SQ + s) * SQ + tid] = e / red[0];
}

// column softmax, stored transposed: P2[b,t,s] = softmax_s(sim[b,s,t])
__global__ void k_softmax_cols(const float* sim, float* P2) {
  int b = blockIdx.y, t = blockIdx.x, tid = threadIdx.x;
  float v = sim[((long)b * SQ + tid) * SQ + t];
  __shared__ float red[SQ];
  red[tid] = v; __syncthreads();
  for (int o = 64; o; o >>= 1) { if (tid < o) red[tid] = fmaxf(red[tid], red[tid + o]); __syncthreads(); }
  float mx = red[0]; __syncthreads();
  float e = expf(v - mx);
  red[tid] = e; __syncthreads();
  for (int o = 64; o; o >>= 1) { if (tid < o) red[tid] += red[tid + o]; __syncthreads(); }
  P2[((long)b * SQ + t) * SQ + tid] = e / red[0];
}

// v[b] = [sum d1, max d1, sum d2, max d2]  (each 600)
__global__ void k_pool(const float* hdec, float* v) {
  int b = blockIdx.x;
  int d = blockIdx.y * blockDim.x + threadIdx.x;   // 0..1199
  if (d >= 1200) return;
  int half = d / 600, j = d - half * 600;
  const float* base = hdec + ((long)(half * BQ + b) * SQ) * 600 + j;
  float sm = 0.0f, mx = -1e30f;
  for (int s = 0; s < SQ; ++s) {
    float x = base[(long)s * 600];
    sm += x;
    mx = fmaxf(mx, x);
  }
  v[(long)b * 2400 + half * 1200 + j]       = sm;
  v[(long)b * 2400 + half * 1200 + 600 + j] = mx;
}

// out = y @ agg_W2 + agg_b2   (64x300 @ 300x3)
__global__ void k_final(const float* y, const float* W2, const float* b2,
                        float* out) {
  int idx = blockIdx.x * blockDim.x + threadIdx.x;
  if (idx >= BQ * NCQ) return;
  int b = idx / NCQ, c = idx - b * NCQ;
  float s = b2[c];
  for (int u = 0; u < DQ; ++u) s += y[b * DQ + u] * W2[u * NCQ + c];
  out[idx] = s;
}

// ------------------------------- orchestration -----------------------------

extern "C" void kernel_launch(void* const* d_in, const int* in_sizes, int n_in,
                              void* d_out, int out_size, void* d_ws,
                              size_t ws_size, hipStream_t stream) {
  (void)in_sizes; (void)n_in; (void)out_size; (void)ws_size;

  const int*   x1       = (const int*)d_in[0];
  const int*   x2       = (const int*)d_in[1];
  const int*   char1    = (const int*)d_in[2];
  const int*   char2    = (const int*)d_in[3];
  // d_in[4], d_in[5] (tag1/tag2) are unused by the reference.
  const int*   temp1    = (const int*)d_in[6];
  const int*   temp2    = (const int*)d_in[7];
  const int*   len1     = (const int*)d_in[8];
  const int*   len2     = (const int*)d_in[9];
  const float* wemb     = (const float*)d_in[10];
  const float* cemb     = (const float*)d_in[11];
  const float* cfilt    = (const float*)d_in[12];
  const float* eWx_f    = (const float*)d_in[13];
  const float* eWh_f    = (const float*)d_in[14];
  const float* eb_f     = (const float*)d_in[15];
  const float* eWx_b    = (const float*)d_in[16];
  const float* eWh_b    = (const float*)d_in[17];
  const float* eb_b     = (const float*)d_in[18];
  const float* dec_W    = (const float*)d_in[19];
  const float* dec_b    = (const float*)d_in[20];
  const float* dWx_f    = (const float*)d_in[21];
  const float* dWh_f    = (const float*)d_in[22];
  const float* db_f     = (const float*)d_in[23];
  const float* dWx_b    = (const float*)d_in[24];
  const float* dWh_b    = (const float*)d_in[25];
  const float* db_b     = (const float*)d_in[26];
  const float* agg_W1   = (const float*)d_in[27];
  const float* agg_b1   = (const float*)d_in[28];
  const float* agg_W2   = (const float*)d_in[29];
  const float* agg_b2   = (const float*)d_in[30];
  float* out = (float*)d_out;

  // ---- workspace carving (floats) ----
  float* ws = (float*)d_ws;
  long off = 0;
  float* tree = ws + off; off += (long)BBQ * SQ * DEQ;   // 6,553,600
  float* henc = ws + off; off += (long)BBQ * SQ * 600;   // 9,830,400
  float* hdec = ws + off; off += (long)BBQ * SQ * 600;
  float* att  = ws + off; off += (long)BBQ * SQ * 600;
  float* pbuf = ws + off; off += (long)BBQ * SQ * DQ;    // 4,915,200
  float* sim  = ws + off; off += (long)BQ * SQ * SQ;     // 1,048,576
  float* P1   = ws + off; off += (long)BQ * SQ * SQ;
  float* P2   = ws + off; off += (long)BQ * SQ * SQ;
  float* Rbuf = ws + off; off += (long)BBQ * H4Q;        // 153,600
  float* hst  = ws + off; off += (long)BBQ * UQ;         // h then c, contiguous
  float* cst  = ws + off; off += (long)BBQ * UQ;
  float* vbuf = ws + off; off += (long)BQ * 2400;
  float* ybuf = ws + off; off += (long)BQ * DQ;

  auto zero = [&](float* p, long n) {
    k_zero<<<(unsigned)((n + 255) / 256), 256, 0, stream>>>(p, n);
  };

  // ---- 1. build padded tree embeddings ----
  zero(tree, (long)BBQ * SQ * DEQ);
  {
    long n = (long)BBQ * LQ * DQ;
    k_embed<<<(unsigned)((n + 255) / 256), 256, 0, stream>>>(x1, x2, wemb, tree);
  }
  k_char<<<BBQ * LQ, 128, 0, stream>>>(char1, char2, cemb, cfilt, tree);
  k_unfold<<<BBQ, 128, 0, stream>>>(temp1, temp2, len1, len2, tree);

  // ---- LSTM pass helper: fused [x_t | h] @ [Wx ; Wh] per step ----
  auto lstm_dir = [&](const float* seqIn, int K1, const float* Wx,
                      const float* Wh, const float* bias, float* seqOut,
                      int colOff, bool reverse) {
    zero(hst, 2L * BBQ * UQ);   // h and c
    for (int s = 0; s < SQ; ++s) {
      int t = reverse ? (SQ - 1 - s) : s;
      GemmP p = {};
      p.A   = seqIn + (long)t * K1;  p.ldA  = (long)SQ * K1; p.sAb = 0;
      p.A2  = hst;                   p.ldA2 = UQ;            p.K1  = K1;
      p.Bm  = Wx;                    p.ldB  = H4Q;           p.sBb = 0;
      p.B2  = Wh;                    p.ldB2 = H4Q;           p.KB1 = K1;
      p.bias = nullptr;
      p.C = Rbuf; p.ldC = H4Q; p.sCb = 0;
      p.M = BBQ; p.N = H4Q; p.K = K1 + UQ;
      run_gemm<1, 1, 0>(stream, p, 1);
      k_cell<<<(BBQ * UQ + 255) / 256, 256, 0, stream>>>(Rbuf, bias, hst, cst,
                                                         seqOut, t, colOff);
    }
  };

  // ---- 2. encoder BiLSTM over tree (both sentences batched, M=128) ----
  lstm_dir(tree, DEQ, eWx_f, eWh_f, eb_f, henc, 0,   false);
  lstm_dir(tree, DEQ, eWx_b, eWh_b, eb_b, henc, 300, true);

  // ---- 3. attention: sim = h1 @ h2^T (batched over B) ----
  {
    GemmP p = {};
    p.A  = henc;                        p.ldA = 600; p.sAb = (long)SQ * 600;
    p.Bm = henc + (long)BQ * SQ * 600;  p.ldB = 600; p.sBb = (long)SQ * 600;
    p.bias = nullptr;
    p.C = sim; p.ldC = SQ; p.sCb = (long)SQ * SQ;
    p.M = SQ; p.N = SQ; p.K = 600;
    run_gemm<0, 2, 0>(stream, p, BQ);   // B transposed
  }
  k_softmax_rows<<<dim3(SQ, BQ), SQ, 0, stream>>>(sim, P1);
  k_softmax_cols<<<dim3(SQ, BQ), SQ, 0, stream>>>(sim, P2);
  {   // beta = P1 @ h2  -> att rows [0, B*S)
    GemmP p = {};
    p.A  = P1;                          p.ldA = SQ;  p.sAb = (long)SQ * SQ;
    p.Bm = henc + (long)BQ * SQ * 600;  p.ldB = 600; p.sBb = (long)SQ * 600;
    p.bias = nullptr;
    p.C = att; p.ldC = 600; p.sCb = (long)SQ * 600;
    p.M = SQ; p.N = 600; p.K = SQ;
    run_gemm<0, 0, 0>(stream, p, BQ);
  }
  {   // alpha = P2 @ h1 -> att rows [B*S, 2B*S)
    GemmP p = {};
    p.A  = P2;   p.ldA = SQ;  p.sAb = (long)SQ * SQ;
    p.Bm = henc; p.ldB = 600; p.sBb = (long)SQ * 600;
    p.bias = nullptr;
    p.C = att + (long)BQ * SQ * 600; p.ldC = 600; p.sCb = (long)SQ * 600;
    p.M = SQ; p.N = 600; p.K = SQ;
    run_gemm<0, 0, 0>(stream, p, BQ);
  }

  // ---- 4. p = relu([h, att, h*att, h-att] @ dec_W + dec_b), m never stored ----
  {
    GemmP p = {};
    p.A  = henc; p.ldA  = 600;
    p.A2 = att;  p.ldA2 = 600; p.K1 = 600;    // segment size for mfuse
    p.Bm = dec_W; p.ldB = DQ; p.sBb = 0;
    p.bias = dec_b;
    p.C = pbuf; p.ldC = DQ; p.sCb = 0;
    p.M = BBQ * SQ; p.N = DQ; p.K = 2400;
    run_gemm<2, 0, 1>(stream, p, 1);          // mfuse A, relu
  }

  // ---- 5. decoder BiLSTM over p ----
  lstm_dir(pbuf, DQ, dWx_f, dWh_f, db_f, hdec, 0,   false);
  lstm_dir(pbuf, DQ, dWx_b, dWh_b, db_b, hdec, 300, true);

  // ---- 6. pool + head ----
  k_pool<<<dim3(BQ, 5), 256, 0, stream>>>(hdec, vbuf);
  {   // y = tanh(v @ agg_W1 + agg_b1)
    GemmP p = {};
    p.A  = vbuf; p.ldA = 2400;
    p.Bm = agg_W1; p.ldB = DQ;
    p.bias = agg_b1;
    p.C = ybuf; p.ldC = DQ;
    p.M = BQ; p.N = DQ; p.K = 2400;
    run_gemm<0, 0, 2>(stream, p, 1);          // tanh
  }
  k_final<<<1, 256, 0, stream>>>(ybuf, agg_W2, agg_b2, out);
}
